// GatedGCNLayer_15968688407203
// MI455X (gfx1250) — compile-verified, matrix operand order checked by reading, not run on hardware
//
#include <hip/hip_runtime.h>
#include <hip/hip_bf16.h>
#include <math.h>

typedef _Float16 v16h __attribute__((ext_vector_type(16)));
typedef _Float16 h8   __attribute__((ext_vector_type(8)));
typedef float    v8f  __attribute__((ext_vector_type(8)));
typedef float    v4f  __attribute__((ext_vector_type(4)));

#define NA 200000
#define NB 300000
#define NG 10000
#define DD 256
#define EPS_GATE 1e-6f
#define EPS_BN   1e-5f

// LDS weight strips, double-buffered: each strip = 32 output columns x 256
// K-halves, padded to 264 halves/column (8-half pad per 256) so 16 lanes
// reading col*264 hit 16 distinct banks. 2 x 8448 halves = 33792 B LDS.
#define SUB_COLS     32
#define COL_STRIDE   264
#define STRIP_ELEMS  (SUB_COLS * DD)            // 8192 f16, contiguous in global
#define BUF_HALVES   (SUB_COLS * COL_STRIDE)    // 8448 halves per buffer

#if __has_builtin(__builtin_amdgcn_tensor_load_to_lds) && __has_builtin(__builtin_amdgcn_s_wait_tensorcnt)
#define USE_TDM 1
#else
#define USE_TDM 0
#endif

// ---------------------------------------------------------------------------
// 0) Convert f32 weights -> f16 (9*256*256*2B = 1.2MB, stays L2-hot)
// ---------------------------------------------------------------------------
__global__ __launch_bounds__(256) void cvt_w16_kernel(const float* __restrict__ W,
                                                      _Float16* __restrict__ w16, int n) {
    int i = blockIdx.x * blockDim.x + threadIdx.x;
    if (i < n) w16[i] = (_Float16)W[i];
}

__global__ __launch_bounds__(256) void zero_kernel(float* __restrict__ p, size_t n) {
    size_t i = (size_t)blockIdx.x * blockDim.x + threadIdx.x;
    size_t stride = (size_t)gridDim.x * blockDim.x;
    for (; i < n; i += stride) p[i] = 0.0f;
}

// ---------------------------------------------------------------------------
// TDM: DMA one contiguous 16KB weight strip into LDS, inserting the bank-
// conflict padding via the D# pad feature (pad_interval=128 DW, pad_amount=4 DW
// => effective column stride 264 halves). Issued by wave 0 only.
// ---------------------------------------------------------------------------
#if USE_TDM
typedef unsigned int v4u __attribute__((ext_vector_type(4)));
typedef int          v8i __attribute__((ext_vector_type(8)));
typedef int          v4i __attribute__((ext_vector_type(4)));
typedef __attribute__((address_space(3))) _Float16 lds_h_t;

__device__ __forceinline__ void tdm_issue_strip(const _Float16* src, _Float16* ldsdst) {
    const unsigned lds_off = (unsigned)(unsigned long long)(lds_h_t*)ldsdst;
    const unsigned long long ga = (unsigned long long)src;

    v4u g0;
    g0.x = 1u;                                      // count=1, user descriptor
    g0.y = lds_off;                                 // LDS byte address
    g0.z = (unsigned)(ga & 0xffffffffu);            // global_addr[31:0]
    g0.w = (unsigned)((ga >> 32) & 0x01ffffffu)     // global_addr[56:32]
         | (2u << 30);                              // type = 2 ("image")

    v8i g1;
    g1[0] = (1 << 16)                               // data_size = 2 bytes
          | (1 << 20)                               // pad_enable
          | (6 << 22)                               // pad_interval: 128 DWORDs (=1 column)
          | (3 << 25);                              // pad_amount:   4 DWORDs (=8 halves)
    g1[1] = (STRIP_ELEMS & 0xffff) << 16;           // tensor_dim0[15:0]
    g1[2] = ((STRIP_ELEMS >> 16) & 0xffff)          // tensor_dim0[31:16]
          | (1 << 16);                              // tensor_dim1 = 1
    g1[3] = (STRIP_ELEMS & 0xffff) << 16;           // tile_dim0 = 8192 (1D tile)
    g1[4] = 0;                                      // tile_dim1/2 unused
    g1[5] = STRIP_ELEMS;                            // tensor_dim0_stride[31:0]
    g1[6] = 0;
    g1[7] = 0;

    v4i gz = {0, 0, 0, 0};
#if defined(__clang_major__) && (__clang_major__ >= 23)
    v8i gz8 = {0, 0, 0, 0, 0, 0, 0, 0};
    __builtin_amdgcn_tensor_load_to_lds(g0, g1, gz, gz, gz8, 0);
#else
    __builtin_amdgcn_tensor_load_to_lds(g0, g1, gz, gz, 0);
#endif
}
#endif

// ---------------------------------------------------------------------------
// 1) Fused multi-matrix GEMM: out[m] = X @ W[mat[m]]^T + b[mat[m]]
//    - one wave per 16-row tile; A fragments for the full K=256 kept in VGPRs
//    - weight strips TDM-DMA'd into double-buffered padded LDS; DMA of strip
//      i+1 overlaps WMMA on strip i (s_wait_tensorcnt 1 pipeline)
//    - accumulator starts at 0; bias added at store time (no pre-WMMA bubble)
// ---------------------------------------------------------------------------
struct GemmJob {
    const float*    X;
    const _Float16* w16;    // base of all 9 f16 weight matrices
    const float*    bias;   // base of all 9 biases (f32)
    float*          out[4];
    int             mat[4];
    int             rows;
    int             nmats;
};

__global__ __launch_bounds__(256) void gemm_wmma_kernel(GemmJob job) {
    __shared__ _Float16 lds_w[2 * BUF_HALVES];   // 33792 B, double-buffered

    const int wave   = threadIdx.x >> 5;
    const int lane   = threadIdx.x & 31;
    const int ntiles = job.rows >> 4;
    int tile = blockIdx.x * 8 + wave;            // 8 waves / 256-thread block
    if (tile >= ntiles) tile = ntiles - 1;       // clamp (duplicate work) so all
                                                 // waves stay alive for barriers
                                                 // and EXEC stays all-1s for WMMA
    const int row_l = lane & 15;                 // A row / B column / C column
    const int hi    = lane >> 4;                 // lane-group select

    // ---- A fragments for the whole K=256 (ISA 16-bit A layout:
    //      element e holds K = (e<8 ? e : e+8) + hi*8 within a 32-wide chunk).
    const float* xrow = job.X + (size_t)(tile * 16 + row_l) * DD;
    v16h afrag[8];
#pragma unroll
    for (int kc = 0; kc < 8; ++kc) {
        const float* p0 = xrow + kc * 32 + hi * 8;
        const float* p1 = p0 + 16;
        v4f x0 = *(const v4f*)(p0);
        v4f x1 = *(const v4f*)(p0 + 4);
        v4f x2 = *(const v4f*)(p1);
        v4f x3 = *(const v4f*)(p1 + 4);
        v16h a;
#pragma unroll
        for (int e = 0; e < 4; ++e) {
            a[e]      = (_Float16)x0[e];
            a[e + 4]  = (_Float16)x1[e];
            a[e + 8]  = (_Float16)x2[e];
            a[e + 12] = (_Float16)x3[e];
        }
        afrag[kc] = a;
    }

    // ---- Strip pipeline over all matrices: strip s -> (mat s>>3, cols (s&7)*32)
    const int nstrips = job.nmats * 8;
    auto strip_ptr = [&](int s) -> const _Float16* {
        return job.w16 + (size_t)job.mat[s >> 3] * DD * DD + (size_t)(s & 7) * STRIP_ELEMS;
    };

    // Prologue: stage strip 0 into buffer 0
#if USE_TDM
    if (threadIdx.x < 32) tdm_issue_strip(strip_ptr(0), &lds_w[0]);
#else
    for (int c = threadIdx.x; c < STRIP_ELEMS / 8; c += 256) {
        const int col = c >> 5, ko = (c & 31) * 8;
        *(h8*)&lds_w[col * COL_STRIDE + ko] = *(const h8*)&strip_ptr(0)[col * DD + ko];
    }
#endif

    for (int s = 0; s < nstrips; ++s) {
        const int m   = s >> 3;
        const int j   = s & 7;
        const int nxt = s + 1;
        _Float16* cur = &lds_w[(s & 1) * BUF_HALVES];

        // L2 prefetch two strips ahead so the next TDM transfer hits L2.
        if (nxt + 1 < nstrips)
            __builtin_prefetch(strip_ptr(nxt + 1) + (size_t)threadIdx.x * 32, 0, 3);

        // Stage strip s+1 into the other buffer; wait until strip s has landed.
#if USE_TDM
        if (threadIdx.x < 32) {
            if (nxt < nstrips) {
                tdm_issue_strip(strip_ptr(nxt), &lds_w[(nxt & 1) * BUF_HALVES]);
                __builtin_amdgcn_s_wait_tensorcnt(1);   // strip s complete, s+1 in flight
            } else {
                __builtin_amdgcn_s_wait_tensorcnt(0);
            }
        }
#else
        if (nxt < nstrips) {
            _Float16* nb = &lds_w[(nxt & 1) * BUF_HALVES];
            const _Float16* np = strip_ptr(nxt);
            for (int c = threadIdx.x; c < STRIP_ELEMS / 8; c += 256) {
                const int col = c >> 5, ko = (c & 31) * 8;
                *(h8*)&nb[col * COL_STRIDE + ko] = *(const h8*)&np[col * DD + ko];
            }
        }
#endif
        __syncthreads();   // strip s visible to all 8 waves

        // Bias for this strip's two N-tiles: issued now, consumed only at store.
        const float* bm  = job.bias + (size_t)job.mat[m] * DD;
        const float  bv0 = bm[j * 32 + 0 * 16 + row_l];
        const float  bv1 = bm[j * 32 + 1 * 16 + row_l];
        float*       om  = job.out[m];

#pragma unroll
        for (int t = 0; t < 2; ++t) {                   // 2 N-tiles per strip
            const int colg = t * 16 + row_l;            // column within strip
            const int col  = j * 32 + colg;             // global output column
            v8f c = {};                                 // zero accum, bias at store

#pragma unroll
            for (int kc = 0; kc < 8; ++kc) {
                // B element (k, n=col) = W[col][k]; same K pattern as A.
                const h8 b0 = *(const h8*)&cur[colg * COL_STRIDE + kc * 32 + hi * 8];
                const h8 b1 = *(const h8*)&cur[colg * COL_STRIDE + kc * 32 + 16 + hi * 8];
                v16h b;
#pragma unroll
                for (int e = 0; e < 8; ++e) { b[e] = b0[e]; b[8 + e] = b1[e]; }
                c = __builtin_amdgcn_wmma_f32_16x16x32_f16(
                        false, afrag[kc], false, b, (short)0, c, false, false);
            }
            // C/D layout: VGPR r -> row (r + hi*8), col fixed per lane.
            const float bv = (t == 0) ? bv0 : bv1;
            float* orow = om + (size_t)(tile * 16 + hi * 8) * DD + col;
#pragma unroll
            for (int r = 0; r < 8; ++r) orow[(size_t)r * DD] = c[r] + bv;
        }
        __syncthreads();   // reads of buffer (s&1) done; safe to overwrite at s+1
    }
}

// ---------------------------------------------------------------------------
// 2) Edge phase: e_new, sigmoid gate, scatter num/den/segHe/cnt_b.
//    e_pre (= e_new * snorm_e) overwrites Be in place.
// ---------------------------------------------------------------------------
__global__ __launch_bounds__(256) void edge_kernel(
        const float* __restrict__ Ah, const float* __restrict__ Eh,
        float* __restrict__ Be, const float* __restrict__ He,
        const float* __restrict__ Cu, const float* __restrict__ snorm_e,
        const int* __restrict__ bond_atoms, const int* __restrict__ bond2graph,
        float* __restrict__ num, float* __restrict__ den,
        float* __restrict__ segHe, float* __restrict__ cnt_b) {
    size_t idx = (size_t)blockIdx.x * blockDim.x + threadIdx.x;
    if (idx >= (size_t)NB * DD) return;
    const int bond = (int)(idx >> 8);
    const int d    = (int)(idx & (DD - 1));
    const int i0   = bond_atoms[2 * bond];
    const int i1   = bond_atoms[2 * bond + 1];
    const int g    = bond2graph[bond];

    const float en  = Ah[(size_t)i0 * DD + d] + Ah[(size_t)i1 * DD + d] +
                      Be[idx] + Cu[(size_t)g * DD + d];
    const float sig = 1.0f / (1.0f + expf(-en));

    atomicAdd(&num[(size_t)i0 * DD + d], sig * Eh[(size_t)i1 * DD + d]);
    atomicAdd(&num[(size_t)i1 * DD + d], sig * Eh[(size_t)i0 * DD + d]);
    atomicAdd(&den[(size_t)i0 * DD + d], sig);
    atomicAdd(&den[(size_t)i1 * DD + d], sig);
    atomicAdd(&segHe[(size_t)g * DD + d], He[idx]);

    Be[idx] = en * snorm_e[bond];           // e_pre (pre-BN)
    if (d == 0) atomicAdd(&cnt_b[g], 1.0f);
}

// ---------------------------------------------------------------------------
// 3) Node phase: h_new; h_pre overwrites Dh in place. Scatter Gh/cnt_a.
// ---------------------------------------------------------------------------
__global__ __launch_bounds__(256) void node_kernel(
        float* __restrict__ Dh, const float* __restrict__ num,
        const float* __restrict__ den, const float* __restrict__ Fu,
        const float* __restrict__ Gh, const float* __restrict__ snorm_n,
        const int* __restrict__ atom2graph,
        float* __restrict__ segGh, float* __restrict__ cnt_a) {
    size_t idx = (size_t)blockIdx.x * blockDim.x + threadIdx.x;
    if (idx >= (size_t)NA * DD) return;
    const int atom = (int)(idx >> 8);
    const int d    = (int)(idx & (DD - 1));
    const int g    = atom2graph[atom];

    const float hn = Dh[idx] + num[idx] / (den[idx] + EPS_GATE) +
                     Fu[(size_t)g * DD + d];
    Dh[idx] = hn * snorm_n[atom];           // h_pre (pre-BN)

    atomicAdd(&segGh[(size_t)g * DD + d], Gh[idx]);
    if (d == 0) atomicAdd(&cnt_a[g], 1.0f);
}

// ---------------------------------------------------------------------------
// 4) Graph phase: u_new; u_pre overwrites Iu in place.
// ---------------------------------------------------------------------------
__global__ __launch_bounds__(256) void graph_kernel(
        float* __restrict__ Iu, const float* __restrict__ segGh,
        const float* __restrict__ segHe, const float* __restrict__ cnt_a,
        const float* __restrict__ cnt_b) {
    size_t idx = (size_t)blockIdx.x * blockDim.x + threadIdx.x;
    if (idx >= (size_t)NG * DD) return;
    const int g = (int)(idx >> 8);
    Iu[idx] = segGh[idx] / fmaxf(cnt_a[g], 1.0f) +
              segHe[idx] / fmaxf(cnt_b[g], 1.0f) + Iu[idx];
}

// ---------------------------------------------------------------------------
// 5) BatchNorm column stats (sum, sumsq per feature) + normalize + ELU.
// ---------------------------------------------------------------------------
__global__ __launch_bounds__(256) void colstats_kernel(
        const float* __restrict__ X, int rows, float* __restrict__ stat) {
    const int d = threadIdx.x;                 // 256 threads = one per column
    float s = 0.0f, ss = 0.0f;
    for (int r = blockIdx.x; r < rows; r += gridDim.x) {
        const float x = X[(size_t)r * DD + d];
        s += x; ss += x * x;
    }
    atomicAdd(&stat[d], s);
    atomicAdd(&stat[DD + d], ss);
}

__global__ __launch_bounds__(256) void bn_elu_kernel(
        const float* __restrict__ X, int rows, const float* __restrict__ stat,
        const float* __restrict__ gamma, const float* __restrict__ beta,
        float* __restrict__ out) {
    size_t idx = (size_t)blockIdx.x * blockDim.x + threadIdx.x;
    if (idx >= (size_t)rows * DD) return;
    const int d    = (int)(idx & (DD - 1));
    const float nv = 1.0f / (float)rows;
    const float m  = stat[d] * nv;
    const float v  = stat[DD + d] * nv - m * m;
    const float y  = gamma[d] * (X[idx] - m) * rsqrtf(v + EPS_BN) + beta[d];
    out[idx] = (y > 0.0f) ? y : expm1f(y);
}

// ---------------------------------------------------------------------------
// Host orchestration
// ---------------------------------------------------------------------------
extern "C" void kernel_launch(void* const* d_in, const int* in_sizes, int n_in,
                              void* d_out, int out_size, void* d_ws, size_t ws_size,
                              hipStream_t stream) {
    const float* h       = (const float*)d_in[0];
    const float* e       = (const float*)d_in[1];
    const float* u       = (const float*)d_in[2];
    const float* snorm_n = (const float*)d_in[3];
    const float* snorm_e = (const float*)d_in[4];
    const float* W       = (const float*)d_in[5];
    const float* b       = (const float*)d_in[6];
    const float* bn_g    = (const float*)d_in[7];
    const float* bn_b    = (const float*)d_in[8];
    const int*   bond_at = (const int*)d_in[9];
    const int*   a2g     = (const int*)d_in[10];
    const int*   b2g     = (const int*)d_in[11];
    float*       out     = (float*)d_out;

    // ---- Workspace carve-up (floats). Total ~474M floats (~1.9 GB).
    float* ws = (float*)d_ws;
    size_t off = 0;
    _Float16* w16 = (_Float16*)ws;          off += (size_t)9 * DD * DD / 2;  // f16 weights
    float* Ah = ws + off;                   off += (size_t)NA * DD;
    float* Dh = ws + off;                   off += (size_t)NA * DD;          // -> h_pre
    float* Eh = ws + off;                   off += (size_t)NA * DD;
    float* Gh = ws + off;                   off += (size_t)NA * DD;
    float* Be = ws + off;                   off += (size_t)NB * DD;          // -> e_pre
    float* He = ws + off;                   off += (size_t)NB * DD;
    float* Cu = ws + off;                   off += (size_t)NG * DD;
    float* Fu = ws + off;                   off += (size_t)NG * DD;
    float* Iu = ws + off;                   off += (size_t)NG * DD;          // -> u_pre
    // ---- zeroed region (contiguous): accumulators + BN stats
    float* zbase = ws + off;
    float* num   = ws + off;                off += (size_t)NA * DD;
    float* den   = ws + off;                off += (size_t)NA * DD;
    float* segGh = ws + off;                off += (size_t)NG * DD;
    float* segHe = ws + off;                off += (size_t)NG * DD;
    float* cnt_a = ws + off;                off += NG;
    float* cnt_b = ws + off;                off += NG;
    float* stat_h = ws + off;               off += 2 * DD;
    float* stat_e = ws + off;               off += 2 * DD;
    float* stat_u = ws + off;               off += 2 * DD;
    const size_t nzero = (size_t)(ws + off - zbase);

    // 0) weights -> f16 ; zero accumulators
    {
        const int n = 9 * DD * DD;
        cvt_w16_kernel<<<(n + 255) / 256, 256, 0, stream>>>(W, w16, n);
        zero_kernel<<<2048, 256, 0, stream>>>(zbase, nzero);
    }

    // 1) The 9 GEMMs as 3 fused WMMA launches (A fragments reused across mats,
    //    weight strips TDM-staged into double-buffered LDS, DMA/compute overlap)
    {
        GemmJob jh; jh.X = h; jh.w16 = w16; jh.bias = b; jh.rows = NA; jh.nmats = 4;
        jh.mat[0] = 0; jh.mat[1] = 3; jh.mat[2] = 4; jh.mat[3] = 6;
        jh.out[0] = Ah; jh.out[1] = Dh; jh.out[2] = Eh; jh.out[3] = Gh;
        const int tiles_h = NA / 16;
        gemm_wmma_kernel<<<(tiles_h + 7) / 8, 256, 0, stream>>>(jh);

        GemmJob je; je.X = e; je.w16 = w16; je.bias = b; je.rows = NB; je.nmats = 2;
        je.mat[0] = 1; je.mat[1] = 7; je.mat[2] = 0; je.mat[3] = 0;
        je.out[0] = Be; je.out[1] = He; je.out[2] = nullptr; je.out[3] = nullptr;
        const int tiles_e = NB / 16;
        gemm_wmma_kernel<<<(tiles_e + 7) / 8, 256, 0, stream>>>(je);

        GemmJob ju; ju.X = u; ju.w16 = w16; ju.bias = b; ju.rows = NG; ju.nmats = 3;
        ju.mat[0] = 2; ju.mat[1] = 5; ju.mat[2] = 8; ju.mat[3] = 0;
        ju.out[0] = Cu; ju.out[1] = Fu; ju.out[2] = Iu; ju.out[3] = nullptr;
        const int tiles_u = NG / 16;
        gemm_wmma_kernel<<<(tiles_u + 7) / 8, 256, 0, stream>>>(ju);
    }

    // 2) edges -> 3) nodes -> 4) graphs
    {
        const size_t ne = (size_t)NB * DD;
        edge_kernel<<<(unsigned)((ne + 255) / 256), 256, 0, stream>>>(
            Ah, Eh, Be, He, Cu, snorm_e, bond_at, b2g, num, den, segHe, cnt_b);

        const size_t nn = (size_t)NA * DD;
        node_kernel<<<(unsigned)((nn + 255) / 256), 256, 0, stream>>>(
            Dh, num, den, Fu, Gh, snorm_n, a2g, segGh, cnt_a);

        const size_t ng = (size_t)NG * DD;
        graph_kernel<<<(unsigned)((ng + 255) / 256), 256, 0, stream>>>(
            Iu, segGh, segHe, cnt_a, cnt_b);
    }

    // 5) BatchNorm + ELU (h_pre in Dh, e_pre in Be, u_pre in Iu)
    {
        colstats_kernel<<<512, 256, 0, stream>>>(Dh, NA, stat_h);
        colstats_kernel<<<512, 256, 0, stream>>>(Be, NB, stat_e);
        colstats_kernel<<<512, 256, 0, stream>>>(Iu, NG, stat_u);

        float* out_h = out;
        float* out_e = out + (size_t)NA * DD;
        float* out_u = out + (size_t)NA * DD + (size_t)NB * DD;
        bn_elu_kernel<<<(unsigned)(((size_t)NA * DD + 255) / 256), 256, 0, stream>>>(
            Dh, NA, stat_h, bn_g + 0 * DD, bn_b + 0 * DD, out_h);
        bn_elu_kernel<<<(unsigned)(((size_t)NB * DD + 255) / 256), 256, 0, stream>>>(
            Be, NB, stat_e, bn_g + 1 * DD, bn_b + 1 * DD, out_e);
        bn_elu_kernel<<<(unsigned)(((size_t)NG * DD + 255) / 256), 256, 0, stream>>>(
            Iu, NG, stat_u, bn_g + 2 * DD, bn_b + 2 * DD, out_u);
    }
}
